// SemanticMapSocialAttentionModel_24206435680312
// MI455X (gfx1250) — matrix operand order repaired and checked by reading, not compiled, not run on hardware
//
#include <hip/hip_runtime.h>
#include <math.h>

#define LHIST 50
#define PSTEPS 30
#define NAG 2048
#define NBLK 32
#define AGB 64           // agents per block

typedef __bf16 bf16_t;
typedef bf16_t v16bf __attribute__((ext_vector_type(16)));
typedef float  v8f   __attribute__((ext_vector_type(8)));

// ---- workspace layout (float slots) ----
static constexpr long WS_BAR    = 0;        // 2 ints: barrier count, generation
static constexpr long WS_POS    = 4;        // 2 floats
static constexpr long WS_PRED   = 6;        // 2 floats
static constexpr long WS_BLKMAX = 8;        // 32
static constexpr long WS_DENOMP = 40;       // 32
static constexpr long WS_HT     = 72;       // 256
static constexpr long WS_TH     = 328;      // 128
static constexpr long WS_TC     = 456;      // 128
static constexpr long WS_HTP    = 1024;     // 32*256 per-block Ht partials
static constexpr long WS_NH     = 9216;     // 2048*256
static constexpr long WS_NC     = WS_NH + 524288;
static constexpr long WS_WPK    = WS_NC + 524288;   // bf16[327680] packed weights

struct Params {
  const float *img, *tpos, *trel, *npos, *nrel;
  const int   *nhist, *ths;
  const float *target_h0, *target_c0;
  const float *W_temb, *b_temb, *Wih_t, *Whh_t, *bih_t, *bhh_t, *W_tatt, *b_tatt;
  const float *nearby_h0, *nearby_c0, *W_nemb, *b_nemb, *Wih_n, *Whh_n, *bih_n, *bhh_n;
  const float *W_natt, *b_natt, *W_pred, *b_pred;
  float *out;
  float *ws;
};

__device__ inline bf16_t f2bf(float f) {
  unsigned u = __float_as_uint(f);
  unsigned r = (u + 0x7FFFu + ((u >> 16) & 1u)) >> 16;   // RNE to bf16
  return __builtin_bit_cast(bf16_t, (unsigned short)r);
}
__device__ inline float sigm(float x) { return 1.f / (1.f + expf(-x)); }

// Sense-reversing grid barrier over NBLK resident blocks.
__device__ inline void grid_sync(float* ws) {
  __syncthreads();
  if (threadIdx.x == 0) {
    int* cnt = (int*)ws;
    int* gen = ((int*)ws) + 1;
    __threadfence();
    int g = __hip_atomic_load(gen, __ATOMIC_ACQUIRE, __HIP_MEMORY_SCOPE_AGENT);
    int prev = __hip_atomic_fetch_add(cnt, 1, __ATOMIC_ACQ_REL, __HIP_MEMORY_SCOPE_AGENT);
    if (prev == NBLK - 1) {
      __hip_atomic_store(cnt, 0, __ATOMIC_RELAXED, __HIP_MEMORY_SCOPE_AGENT);
      __hip_atomic_fetch_add(gen, 1, __ATOMIC_RELEASE, __HIP_MEMORY_SCOPE_AGENT);
    } else {
      while (__hip_atomic_load(gen, __ATOMIC_ACQUIRE, __HIP_MEMORY_SCOPE_AGENT) == g)
        __builtin_amdgcn_s_sleep(1);
    }
    __threadfence();
  }
  __syncthreads();
}

// Target LSTM (NH=128), run uniformly by all 256 threads of block 0.
__device__ inline void target_lstm(const Params& P, float* ws, const float* xt,
                                   float* thn, float* tcn, bool tmask) {
  const int tid = threadIdx.x;
  float* th = ws + WS_TH;
  float* tc = ws + WS_TC;
  if (tid < 128) {
    const int u = tid;
    float gi = P.bih_t[u]       + P.bhh_t[u];
    float gf = P.bih_t[128 + u] + P.bhh_t[128 + u];
    float gg = P.bih_t[256 + u] + P.bhh_t[256 + u];
    float go = P.bih_t[384 + u] + P.bhh_t[384 + u];
    for (int k = 0; k < 64; ++k) {
      const float x = xt[k]; const float* w = P.Wih_t + (size_t)k * 512;
      gi += x * w[u]; gf += x * w[128 + u]; gg += x * w[256 + u]; go += x * w[384 + u];
    }
    for (int k = 0; k < 128; ++k) {
      const float h = th[k]; const float* w = P.Whh_t + (size_t)k * 512;
      gi += h * w[u]; gf += h * w[128 + u]; gg += h * w[256 + u]; go += h * w[384 + u];
    }
    const float c2 = sigm(gf) * tc[u] + sigm(gi) * tanhf(gg);
    const float h2 = sigm(go) * tanhf(c2);
    thn[u] = tmask ? h2 : th[u];
    tcn[u] = tmask ? c2 : tc[u];
  }
  __syncthreads();
  if (tid < 128) { th[tid] = thn[tid]; tc[tid] = tcn[tid]; }
  __syncthreads();
}

// Pack [Wih_n; Whh_n] (320x1024 f32) into bf16 WMMA-B-lane order; init state.
__global__ __launch_bounds__(256) void social_prepack(Params P) {
  float* ws = P.ws;
  const long gid = (long)blockIdx.x * blockDim.x + threadIdx.x;
  const long stride = (long)gridDim.x * blockDim.x;
  bf16_t* Wpk = (bf16_t*)(ws + WS_WPK);
  // B tile (kt, nt): lane holds K = i + 16*(lane>>4), N = nt*16 + (lane&15)
  for (long idx = gid; idx < 10L * 64 * 512; idx += stride) {
    const int i    = (int)(idx & 15);
    const int lane = (int)((idx >> 4) & 31);
    const long tile = idx >> 9;
    const int kt = (int)(tile >> 6);
    const int nt = (int)(tile & 63);
    const int k   = kt * 32 + i + 16 * (lane >> 4);
    const int col = nt * 16 + (lane & 15);
    const float v = (k < 64) ? P.Wih_n[(size_t)k * 1024 + col]
                             : P.Whh_n[(size_t)(k - 64) * 1024 + col];
    Wpk[idx] = f2bf(v);
  }
  float* nh = ws + WS_NH; float* nc = ws + WS_NC;
  for (long idx = gid; idx < (long)NAG * 256; idx += stride) {
    const int u = (int)(idx & 255);
    nh[idx] = P.nearby_h0[u];
    nc[idx] = P.nearby_c0[u];
  }
  for (long u = gid; u < 256; u += stride) ws[WS_HT + u] = P.nearby_h0[u];
  for (long u = gid; u < 128; u += stride) { ws[WS_TH + u] = P.target_h0[u]; ws[WS_TC + u] = P.target_c0[u]; }
  if (gid == 0) {
    ((int*)ws)[0] = 0; ((int*)ws)[1] = 0;
    ws[WS_POS]     = P.tpos[(LHIST - 1) * 2 + 0];
    ws[WS_POS + 1] = P.tpos[(LHIST - 1) * 2 + 1];
    ws[WS_PRED] = 0.f; ws[WS_PRED + 1] = 0.f;
  }
}

__global__ __launch_bounds__(256) void social_main(Params P) {
  float* ws = P.ws;
  const int tid  = threadIdx.x;
  const int bid  = blockIdx.x;
  const int base = bid * AGB;
  const int lane = tid & 31;
  const int wv   = tid >> 5;

  __shared__ bf16_t zs[AGB * 320];     // [x_emb | h] staged in bf16 (40KB)
  __shared__ float sc[64], wl[64], atl[64], xt[64];
  __shared__ int   mskl[64];
  __shared__ float thn[128], tcn[128];
  __shared__ float red0[256], red1[256];
  __shared__ float s_scal[4];          // 0=currN 1=gmax 2=qx 3=qy

  float* nhg = ws + WS_NH;
  float* ncg = ws + WS_NC;
  const bf16_t* Wpk = (const bf16_t*)(ws + WS_WPK);
  const int ths = P.ths[0];

  for (int step = 0; step < LHIST - 1 + PSTEPS; ++step) {
    const bool isObs = step < LHIST - 1;
    const int t     = isObs ? (step + 1) : (LHIST - 1);
    const int pstep = isObs ? 0 : (step - (LHIST - 1));

    // ---- block 0: prediction head + target LSTM (serial, tiny) ----
    // Cross-step ordering is guaranteed by the previous step's D barrier:
    // block 0 only reads its own ws slots (TH/TC/HT/POS) here.
    if (bid == 0) {
      if (!isObs) {
        float p0 = 0.f, p1 = 0.f;
        for (int k = tid; k < 2432; k += 256) {
          const float v = (k < 128) ? ws[WS_TH + k]
                        : (k < 2176) ? P.img[k - 128]
                                     : ws[WS_HT + (k - 2176)];
          p0 += v * P.W_pred[2 * k];
          p1 += v * P.W_pred[2 * k + 1];
        }
        red0[tid] = p0; red1[tid] = p1; __syncthreads();
        for (int s = 128; s > 0; s >>= 1) {
          if (tid < s) { red0[tid] += red0[tid + s]; red1[tid] += red1[tid + s]; }
          __syncthreads();
        }
        if (tid == 0) {
          const float q0 = red0[0] + P.b_pred[0], q1 = red1[0] + P.b_pred[1];
          ws[WS_PRED] = q0; ws[WS_PRED + 1] = q1;
          ws[WS_POS] += q0; ws[WS_POS + 1] += q1;
          P.out[pstep * 2 + 0] = q0; P.out[pstep * 2 + 1] = q1;
        }
        __syncthreads();
        const float q0 = ws[WS_PRED], q1 = ws[WS_PRED + 1];
        if (tid < 64) xt[tid] = q0 * P.W_temb[tid] + q1 * P.W_temb[64 + tid] + P.b_temb[tid];
        __syncthreads();
        target_lstm(P, ws, xt, thn, tcn, true);
      } else {
        const float r0 = P.trel[t * 2 + 0], r1 = P.trel[t * 2 + 1];
        if (tid < 64) xt[tid] = r0 * P.W_temb[tid] + r1 * P.W_temb[64 + tid] + P.b_temb[tid];
        __syncthreads();
        target_lstm(P, ws, xt, thn, tcn, ths > LHIST - t);
      }
    }

    // ---- heavy: nearby LSTM gate GEMM via bf16 WMMA ----
    // Loop nest keyed for B reuse: each wave owns 2 unit-groups j; for each
    // (j,kt,gate) the B tile is loaded ONCE and multiplied against all 4
    // M-tiles (A held in registers) -> minimal L2 weight traffic.
    {
      const int tn = isObs ? t : (LHIST - 1);
      for (int idx = tid; idx < AGB * 64; idx += 256) {        // x embedding
        const int m = idx >> 6, e = idx & 63, n = base + m;
        const float a = P.nrel[(n * LHIST + tn) * 2 + 0];
        const float b = P.nrel[(n * LHIST + tn) * 2 + 1];
        zs[m * 320 + e] = f2bf(a * P.W_nemb[e] + b * P.W_nemb[64 + e] + P.b_nemb[e]);
      }
      for (int idx = tid; idx < AGB * 256; idx += 256) {       // h (pre-update)
        const int m = idx >> 8, u = idx & 255;
        zs[m * 320 + 64 + u] = f2bf(nhg[(size_t)(base + m) * 256 + u]);
      }
      __syncthreads();

      const int jbase = wv * 2;
      const int lhalf = lane >> 4;
      const int lmod  = lane & 15;
      for (int jj = 0; jj < 2; ++jj) {
        const int j = jbase + jj;
        v8f acc[4][4];                 // [gate][mtile]
        #pragma unroll
        for (int g = 0; g < 4; ++g)
          #pragma unroll
          for (int m = 0; m < 4; ++m)
            #pragma unroll
            for (int r = 0; r < 8; ++r) acc[g][m][r] = 0.f;

        for (int kt = 0; kt < 10; ++kt) {
          v16bf a[4];
          const int kb = kt * 32 + 16 * lhalf;
          #pragma unroll
          for (int m = 0; m < 4; ++m) {
            const int row = m * 16 + lmod;
            #pragma unroll
            for (int i = 0; i < 16; ++i)                        // contiguous k = 16*lhalf + i (matches packed B)
              a[m][i] = zs[row * 320 + kb + i];
          }
          #pragma unroll
          for (int g = 0; g < 4; ++g) {
            const int nt = g * 16 + j;
            const bf16_t* bp = Wpk + ((size_t)(kt * 64 + nt) * 512) + lane * 16;
            v16bf b;
            #pragma unroll
            for (int i = 0; i < 16; ++i) b[i] = bp[i];
            #pragma unroll
            for (int m = 0; m < 4; ++m)
              acc[g][m] = __builtin_amdgcn_wmma_f32_16x16x32_bf16(
                  false, a[m], false, b, (short)0, acc[g][m], false, false);
          }
        }

        const int u = j * 16 + lmod;
        const float bi  = P.bih_n[u]       + P.bhh_n[u];
        const float bff = P.bih_n[256 + u] + P.bhh_n[256 + u];
        const float bg  = P.bih_n[512 + u] + P.bhh_n[512 + u];
        const float bo  = P.bih_n[768 + u] + P.bhh_n[768 + u];
        #pragma unroll
        for (int m = 0; m < 4; ++m) {
          #pragma unroll
          for (int r = 0; r < 8; ++r) {
            const int n = base + m * 16 + r + 8 * lhalf;        // ISA C/D layout
            const size_t off = (size_t)n * 256 + u;
            const bool mk = isObs ? (P.nhist[n] > LHIST - t) : true;
            const float hold = nhg[off], cold = ncg[off];
            const float ig = sigm(acc[0][m][r] + bi);
            const float fg = sigm(acc[1][m][r] + bff);
            const float gg = tanhf(acc[2][m][r] + bg);
            const float og = sigm(acc[3][m][r] + bo);
            const float c2 = fg * cold + ig * gg;
            const float h2 = og * tanhf(c2);
            nhg[off] = mk ? h2 : hold;
            ncg[off] = mk ? c2 : cold;
          }
        }
      }
    }

    // Only the prediction phase needs a barrier here: other blocks must see
    // block 0's pred/pos before computing attention scores. In the obs phase
    // the query comes straight from the inputs.
    if (!isObs) grid_sync(ws);   // B

    // ---- attention: query, currN, scores ----
    if (tid < 64) {
      float a0, a1;
      if (isObs) { a0 = P.trel[t * 2]; a1 = P.trel[t * 2 + 1]; }
      else       { a0 = ws[WS_PRED];   a1 = ws[WS_PRED + 1]; }
      atl[tid] = a0 * P.W_tatt[tid] + a1 * P.W_tatt[64 + tid] + P.b_tatt[tid];
    }
    if (tid == 0) {
      if (isObs) { s_scal[2] = P.tpos[t * 2]; s_scal[3] = P.tpos[t * 2 + 1]; }
      else       { s_scal[2] = ws[WS_POS];    s_scal[3] = ws[WS_POS + 1]; }
    }
    if (isObs) {
      int c = 0;
      for (int i = tid; i < NAG; i += 256) c += (P.nhist[i] > LHIST - t) ? 1 : 0;
      red0[tid] = (float)c; __syncthreads();
      for (int s = 128; s > 0; s >>= 1) { if (tid < s) red0[tid] += red0[tid + s]; __syncthreads(); }
      if (tid == 0) s_scal[0] = red0[0];
    } else if (tid == 0) s_scal[0] = (float)NAG;
    __syncthreads();

    if (tid < 64) {
      const int m = tid, n = base + m;
      const int tn = isObs ? t : (LHIST - 1);
      const float dx = s_scal[2] - P.npos[(n * LHIST + tn) * 2 + 0];
      const float dy = s_scal[3] - P.npos[(n * LHIST + tn) * 2 + 1];
      float s = 0.f;
      for (int e = 0; e < 64; ++e)
        s += (dx * P.W_natt[e] + dy * P.W_natt[64 + e] + P.b_natt[e]) * atl[e];
      sc[m] = s * s_scal[0] * 0.125f;   // * currN / sqrt(64)
      mskl[m] = isObs ? ((P.nhist[n] > LHIST - t) ? 1 : 0) : 1;
    }
    __syncthreads();
    if (tid == 0) {
      float mx = -INFINITY;
      for (int m = 0; m < 64; ++m) if (mskl[m]) mx = fmaxf(mx, sc[m]);
      ws[WS_BLKMAX + bid] = mx;
    }

    grid_sync(ws);   // C: block maxima ready (also orders own nh for HtP pass)

    if (tid == 0) {
      float mx = -INFINITY;
      for (int b = 0; b < NBLK; ++b) mx = fmaxf(mx, ws[WS_BLKMAX + b]);
      s_scal[1] = mx;
    }
    __syncthreads();
    if (tid < 64) wl[tid] = mskl[tid] ? expf(sc[tid] - s_scal[1]) : 0.f;
    __syncthreads();
    if (tid == 0) {
      float d = 0.f;
      for (int m = 0; m < 64; ++m) d += wl[m];
      ws[WS_DENOMP + bid] = d;
    }
    {
      float s = 0.f;
      const float* nb = nhg + (size_t)base * 256;
      for (int m = 0; m < 64; ++m) s += wl[m] * nb[m * 256 + tid];
      ws[WS_HTP + bid * 256 + tid] = s;
    }

    grid_sync(ws);   // D: partial sums ready

    if (bid == 0 && s_scal[0] >= 1.0f) {
      float den = 0.f, s = 0.f;
      for (int b = 0; b < NBLK; ++b) { den += ws[WS_DENOMP + b]; s += ws[WS_HTP + b * 256 + tid]; }
      ws[WS_HT + tid] = s / den;
    }
  }
}

extern "C" void kernel_launch(void* const* d_in, const int* in_sizes, int n_in,
                              void* d_out, int out_size, void* d_ws, size_t ws_size,
                              hipStream_t stream) {
  (void)in_sizes; (void)n_in; (void)out_size; (void)ws_size;
  Params P;
  P.img       = (const float*)d_in[0];
  P.tpos      = (const float*)d_in[1];
  P.trel      = (const float*)d_in[2];
  P.npos      = (const float*)d_in[3];
  P.nrel      = (const float*)d_in[4];
  P.nhist     = (const int*)d_in[5];
  P.ths       = (const int*)d_in[6];
  P.target_h0 = (const float*)d_in[7];
  P.target_c0 = (const float*)d_in[8];
  P.W_temb = (const float*)d_in[9];  P.b_temb = (const float*)d_in[10];
  P.Wih_t  = (const float*)d_in[11]; P.Whh_t  = (const float*)d_in[12];
  P.bih_t  = (const float*)d_in[13]; P.bhh_t  = (const float*)d_in[14];
  P.W_tatt = (const float*)d_in[15]; P.b_tatt = (const float*)d_in[16];
  P.nearby_h0 = (const float*)d_in[17]; P.nearby_c0 = (const float*)d_in[18];
  P.W_nemb = (const float*)d_in[19]; P.b_nemb = (const float*)d_in[20];
  P.Wih_n  = (const float*)d_in[21]; P.Whh_n  = (const float*)d_in[22];
  P.bih_n  = (const float*)d_in[23]; P.bhh_n  = (const float*)d_in[24];
  P.W_natt = (const float*)d_in[25]; P.b_natt = (const float*)d_in[26];
  P.W_pred = (const float*)d_in[27]; P.b_pred = (const float*)d_in[28];
  P.out = (float*)d_out;
  P.ws  = (float*)d_ws;

  social_prepack<<<dim3(1024), dim3(256), 0, stream>>>(P);
  social_main<<<dim3(NBLK), dim3(256), 0, stream>>>(P);
}
